// Qwen3MoeSparseMoeBlock_90984587198973
// MI455X (gfx1250) — matrix operand order repaired and checked by reading, not compiled, not run on hardware
//
#include <hip/hip_runtime.h>
#include <math.h>

// ---------------- problem constants ----------------
#define T_TOK 2048
#define HDIM  1024
#define NEXP  16
#define NDIM  512        // moe_intermediate_size
#define TOPK  4
#define TILE  64         // tokens per expert tile
#define XS    1032       // padded LDS stride for x tile (bf16 elems)  -> bank-conflict free
#define H2S   520        // padded LDS stride for h2 tile (bf16 elems)

typedef __bf16  bf16_t;
typedef __bf16  v16bf __attribute__((ext_vector_type(16)));
typedef __bf16  v8bf  __attribute__((ext_vector_type(8)));
typedef float   v8f   __attribute__((ext_vector_type(8)));
typedef float   v4f   __attribute__((ext_vector_type(4)));

static __device__ __forceinline__ v8f vzero8() {
    v8f z;
#pragma unroll
    for (int i = 0; i < 8; ++i) z[i] = 0.0f;
    return z;
}

// 16 consecutive f32 -> v16bf (B fragment for one lane), via b128 loads
static __device__ __forceinline__ v16bf cvt16_f32_bf16(const float* __restrict__ p) {
    v4f f0 = ((const v4f*)p)[0];
    v4f f1 = ((const v4f*)p)[1];
    v4f f2 = ((const v4f*)p)[2];
    v4f f3 = ((const v4f*)p)[3];
    v16bf r;
#pragma unroll
    for (int i = 0; i < 4; ++i) r[i]      = (bf16_t)f0[i];
#pragma unroll
    for (int i = 0; i < 4; ++i) r[4 + i]  = (bf16_t)f1[i];
#pragma unroll
    for (int i = 0; i < 4; ++i) r[8 + i]  = (bf16_t)f2[i];
#pragma unroll
    for (int i = 0; i < 4; ++i) r[12 + i] = (bf16_t)f3[i];
    return r;
}

// A fragment (16x32 bf16) from a row-major bf16 LDS tile with stride S.
// k order used here (contiguous, same as B): lane half=l>>4, M=l&15; elems 0..15 -> K=k0+half*16+i
// (two 16B ds loads)
static __device__ __forceinline__ v16bf load_a_frag(const bf16_t* __restrict__ base,
                                                    int S, int rowbase, int k0, int lane) {
    const int half = lane >> 4;
    const int m    = lane & 15;
    const bf16_t* p = base + (size_t)(rowbase + m) * S + k0 + half * 16;
    v8bf lo = *(const v8bf*)p;
    v8bf hi = *(const v8bf*)(p + 8);
    return __builtin_shufflevector(lo, hi, 0,1,2,3,4,5,6,7,8,9,10,11,12,13,14,15);
}

// ---------------- kernel 0: zero the combined-output accumulator ----------------
__global__ void k_zero(float* __restrict__ p, int n) {
    int i = blockIdx.x * blockDim.x + threadIdx.x;
    int stride = gridDim.x * blockDim.x;
    for (; i < n; i += stride) p[i] = 0.0f;
}

// ---------------- kernel 1: router (logits + softmax + top4 + renorm) ----------------
__global__ void k_router(const float* __restrict__ x,        // [T,H]
                         const float* __restrict__ gate_w,   // [E,H]
                         float* __restrict__ logits_out,     // [T,E]
                         int*   __restrict__ topk_ids,       // [T,4]
                         float* __restrict__ topk_w) {       // [T,4]
    extern __shared__ float sgate[];                          // E*H f32 = 64KB
    const int tid = threadIdx.x;
    for (int i = tid; i < NEXP * HDIM; i += blockDim.x) sgate[i] = gate_w[i];
    __syncthreads();

    const int token = blockIdx.x * blockDim.x + tid;
    if (token >= T_TOK) return;

    float acc[NEXP];
#pragma unroll
    for (int e = 0; e < NEXP; ++e) acc[e] = 0.0f;

    const float* xr = x + (size_t)token * HDIM;
    for (int k = 0; k < HDIM; ++k) {
        float xv = xr[k];
#pragma unroll
        for (int e = 0; e < NEXP; ++e) acc[e] = fmaf(xv, sgate[e * HDIM + k], acc[e]);
    }

    // raw logits out
#pragma unroll
    for (int e = 0; e < NEXP; ++e) logits_out[(size_t)token * NEXP + e] = acc[e];

    // softmax
    float mx = acc[0];
#pragma unroll
    for (int e = 1; e < NEXP; ++e) mx = fmaxf(mx, acc[e]);
    float sum = 0.0f;
    float p[NEXP];
#pragma unroll
    for (int e = 0; e < NEXP; ++e) { p[e] = __expf(acc[e] - mx); sum += p[e]; }
    float inv = __builtin_amdgcn_rcpf(sum);
#pragma unroll
    for (int e = 0; e < NEXP; ++e) p[e] *= inv;

    // top-4 (first-occurrence on ties, like lax.top_k)
    int   ids[TOPK];
    float ws[TOPK];
    bool used[NEXP];
#pragma unroll
    for (int e = 0; e < NEXP; ++e) used[e] = false;
    float s4 = 0.0f;
#pragma unroll
    for (int j = 0; j < TOPK; ++j) {
        float best = -1.0f; int bi = 0;
#pragma unroll
        for (int e = 0; e < NEXP; ++e)
            if (!used[e] && p[e] > best) { best = p[e]; bi = e; }
        used[bi] = true; ids[j] = bi; ws[j] = best; s4 += best;
    }
    float r = __builtin_amdgcn_rcpf(s4);   // NORM_TOPK
#pragma unroll
    for (int j = 0; j < TOPK; ++j) {
        topk_ids[token * TOPK + j] = ids[j];
        topk_w[token * TOPK + j]   = ws[j] * r;
    }
}

// ---------------- kernel 2: deterministic per-expert compaction ----------------
__global__ void k_compact(const int* __restrict__ topk_ids, const float* __restrict__ topk_w,
                          int* __restrict__ counts, int* __restrict__ tok_ids,
                          float* __restrict__ tok_w) {
    const int e = threadIdx.x;
    if (e >= NEXP) return;
    int cnt = 0;
    for (int t = 0; t < T_TOK; ++t) {
#pragma unroll
        for (int j = 0; j < TOPK; ++j) {
            if (topk_ids[t * TOPK + j] == e) {
                tok_ids[e * T_TOK + cnt] = t;
                tok_w[e * T_TOK + cnt]   = topk_w[t * TOPK + j];
                ++cnt;
            }
        }
    }
    counts[e] = cnt;
}

// ---------------- kernel 3: fused expert (w1 GEMM + SiLU*up + w2 GEMM + scatter) --------
// grid = (NEXP, T_TOK/TILE), block = 256 (8 waves)
__global__ void __launch_bounds__(256, 1)
k_expert(const float* __restrict__ x,    // [T,H]
         const float* __restrict__ w1,   // [E,2N,H]
         const float* __restrict__ w2,   // [E,H,N]
         const int*   __restrict__ counts,
         const int*   __restrict__ tok_ids,
         const float* __restrict__ tok_w,
         float* __restrict__ out) {      // [T,H] accumulator (pre-zeroed)
    extern __shared__ char smem[];
    bf16_t* xl   = (bf16_t*)smem;                                     // TILE x XS
    bf16_t* h2l  = (bf16_t*)(smem + (size_t)TILE * XS * 2);           // TILE x H2S
    int*    stok = (int*)  (smem + (size_t)TILE * XS * 2 + (size_t)TILE * H2S * 2);
    float*  sw   = (float*)(stok + TILE);

    const int e  = blockIdx.x;
    const int tb = blockIdx.y * TILE;
    const int cnt = counts[e];
    if (tb >= cnt) return;
    const int nt = min(TILE, cnt - tb);

    const int tid  = threadIdx.x;
    const int wid  = tid >> 5;
    const int lane = tid & 31;
    const int half = lane >> 4;
    const int nl   = lane & 15;

    // ---- token ids / routing weights ----
    if (tid < TILE) {
        if (tid < nt) {
            stok[tid] = tok_ids[e * T_TOK + tb + tid];
            sw[tid]   = tok_w[e * T_TOK + tb + tid];
        } else {
            stok[tid] = 0;
            sw[tid]   = 0.0f;
        }
    }
    __syncthreads();

    // ---- stage A: gather x rows -> bf16 LDS tile ----
    for (int u = tid; u < TILE * (HDIM / 8); u += 256) {
        const int row = u >> 7;            // HDIM/8 = 128 units per row
        const int col = (u & 127) * 8;
        v8bf o;
        if (row < nt) {
            const float* px = x + (size_t)stok[row] * HDIM + col;
            v4f a0 = ((const v4f*)px)[0];
            v4f a1 = ((const v4f*)px)[1];
#pragma unroll
            for (int i = 0; i < 4; ++i) o[i]     = (bf16_t)a0[i];
#pragma unroll
            for (int i = 0; i < 4; ++i) o[4 + i] = (bf16_t)a1[i];
        } else {
#pragma unroll
            for (int i = 0; i < 8; ++i) o[i] = (bf16_t)0.0f;
        }
        *(v8bf*)(xl + (size_t)row * XS + col) = o;
    }
    __syncthreads();

    // ---- stage B: h = x @ w1^T, fused SiLU(gate)*up -> h2 LDS tile ----
    const float* w1e = w1 + (size_t)e * (2 * NDIM) * HDIM;
#pragma unroll 1
    for (int ci = 0; ci < 4; ++ci) {
        const int c = wid + ci * 8;        // n-chunk 0..31 (16 cols each)
        v8f accg[4], accu[4];
#pragma unroll
        for (int m = 0; m < 4; ++m) { accg[m] = vzero8(); accu[m] = vzero8(); }

        const float* pg = w1e + (size_t)(c * 16 + nl) * HDIM + half * 16;             // gate rows
        const float* pu = w1e + (size_t)(NDIM + c * 16 + nl) * HDIM + half * 16;      // up rows

#pragma unroll 1
        for (int k0 = 0; k0 < HDIM; k0 += 32) {
            v16bf bg = cvt16_f32_bf16(pg + k0);
            v16bf bu = cvt16_f32_bf16(pu + k0);
#pragma unroll
            for (int m = 0; m < 4; ++m) {
                v16bf a = load_a_frag(xl, XS, m * 16, k0, lane);
                accg[m] = __builtin_amdgcn_wmma_f32_16x16x32_bf16(
                    false, a, false, bg, (short)0, accg[m], false, false);
                accu[m] = __builtin_amdgcn_wmma_f32_16x16x32_bf16(
                    false, a, false, bu, (short)0, accu[m], false, false);
            }
        }
        // SiLU(gate)*up -> h2 (D layout: M = r + half*8, N = nl)
        // sigmoid via single v_rcp_f32 (denominator >= 1, no edge cases) instead of
        // the IEEE divide expansion (v_div_scale/v_div_fmas chains).
#pragma unroll
        for (int m = 0; m < 4; ++m) {
#pragma unroll
            for (int r = 0; r < 8; ++r) {
                const float g  = accg[m][r];
                const float u  = accu[m][r];
                const float sg = __builtin_amdgcn_rcpf(1.0f + __expf(-g));
                const float v  = g * sg * u;
                h2l[(size_t)(m * 16 + half * 8 + r) * H2S + c * 16 + nl] = (bf16_t)v;
            }
        }
    }
    __syncthreads();

    // ---- stage C: y = h2 @ w2^T, scale by routing weight, scatter-add ----
    const float* w2e = w2 + (size_t)e * HDIM * NDIM;
#pragma unroll 1
    for (int hi = 0; hi < 8; ++hi) {
        const int hc = wid + hi * 8;       // h-chunk 0..63 (16 cols each)
        v8f acc[4];
#pragma unroll
        for (int m = 0; m < 4; ++m) acc[m] = vzero8();

        const float* pb = w2e + (size_t)(hc * 16 + nl) * NDIM + half * 16;
#pragma unroll 1
        for (int k0 = 0; k0 < NDIM; k0 += 32) {
            v16bf b = cvt16_f32_bf16(pb + k0);
#pragma unroll
            for (int m = 0; m < 4; ++m) {
                v16bf a = load_a_frag(h2l, H2S, m * 16, k0, lane);
                acc[m] = __builtin_amdgcn_wmma_f32_16x16x32_bf16(
                    false, a, false, b, (short)0, acc[m], false, false);
            }
        }
#pragma unroll
        for (int m = 0; m < 4; ++m) {
#pragma unroll
            for (int r = 0; r < 8; ++r) {
                const int row = m * 16 + half * 8 + r;
                if (row < nt) {
                    unsafeAtomicAdd(out + (size_t)stok[row] * HDIM + hc * 16 + nl,
                                    acc[m][r] * sw[row]);
                }
            }
        }
    }
}

// ---------------- launch ----------------
extern "C" void kernel_launch(void* const* d_in, const int* in_sizes, int n_in,
                              void* d_out, int out_size, void* d_ws, size_t ws_size,
                              hipStream_t stream) {
    const float* x      = (const float*)d_in[0];   // [1,2048,1024]
    const float* gate_w = (const float*)d_in[1];   // [16,1024]
    const float* w1     = (const float*)d_in[2];   // [16,1024,1024]
    const float* w2     = (const float*)d_in[3];   // [16,1024,512]

    float* out        = (float*)d_out;                         // [T,H]
    float* logits_out = out + (size_t)T_TOK * HDIM;            // [T,E]

    // workspace layout
    char* ws = (char*)d_ws;
    int*   counts   = (int*)ws;                                          // 16
    int*   topk_ids = (int*)(ws + 64);                                   // T*4
    float* topk_w   = (float*)(ws + 64 + T_TOK * TOPK * 4);              // T*4
    int*   tok_ids  = (int*)(ws + 64 + 2 * T_TOK * TOPK * 4);            // E*T
    float* tok_w    = (float*)(ws + 64 + 2 * T_TOK * TOPK * 4 + NEXP * T_TOK * 4);

    // 0) zero the output accumulator
    k_zero<<<512, 256, 0, stream>>>(out, T_TOK * HDIM);

    // 1) router: logits + softmax + top4 + renorm
    k_router<<<T_TOK / 256, 256, NEXP * HDIM * sizeof(float), stream>>>(
        x, gate_w, logits_out, topk_ids, topk_w);

    // 2) deterministic per-expert compaction
    k_compact<<<1, 32, 0, stream>>>(topk_ids, topk_w, counts, tok_ids, tok_w);

    // 3) fused expert compute (WMMA bf16, f32 accumulate)
    const size_t smem3 = (size_t)TILE * XS * 2 + (size_t)TILE * H2S * 2 + (size_t)TILE * 8;
    dim3 g3(NEXP, T_TOK / TILE);
    k_expert<<<g3, 256, smem3, stream>>>(x, w1, w2, counts, tok_ids, tok_w, out);
}
